// EntailmentSelfAttention_30399778521397
// MI455X (gfx1250) — matrix operand order repaired and checked by reading, not compiled, not run on hardware
//
#include <hip/hip_runtime.h>
#include <hip/hip_bf16.h>

typedef __attribute__((ext_vector_type(16))) _Float16 v16h;
typedef __attribute__((ext_vector_type(8)))  _Float16 v8h;
typedef __attribute__((ext_vector_type(8)))  float    v8f;
typedef __attribute__((ext_vector_type(4)))  unsigned int v4u;
typedef __attribute__((ext_vector_type(4)))  int      v4i;
typedef __attribute__((ext_vector_type(8)))  int      v8i;

#define WMMA_F16(a, b, c) \
  __builtin_amdgcn_wmma_f32_16x16x32_f16(false, (a), false, (b), (short)0, (c), false, false)

#if defined(__gfx1250__) && defined(__has_builtin)
#if __has_builtin(__builtin_amdgcn_tensor_load_to_lds) && \
    __has_builtin(__builtin_amdgcn_s_wait_tensorcnt)
#define HAVE_TDM 1
#endif
#endif

// ---- fragment loaders (wave32, 16x16x32 f16 WMMA) -------------------------
// A (16x32, MxK): lane m = lane&15; halves e -> K = (e>>3)*16 + (lane>>4)*8 + (e&7)
__device__ __forceinline__ v16h load_frag_a(const _Float16* base, int stride) {
  const int lane = threadIdx.x & 31;
  const int m  = lane & 15;
  const int hi = (lane >> 4) * 8;
  const _Float16* r = base + m * stride + hi;
  v8h lo = *(const v8h*)(r);
  v8h hh = *(const v8h*)(r + 16);
  v16h out;
#pragma unroll
  for (int i = 0; i < 8; ++i) { out[i] = lo[i]; out[i + 8] = hh[i]; }
  return out;
}

// A fragment from f32 LDS (attention probabilities), converted to f16
__device__ __forceinline__ v16h load_frag_a_f32(const float* base, int stride) {
  const int lane = threadIdx.x & 31;
  const int m  = lane & 15;
  const int hi = (lane >> 4) * 8;
  const float* r = base + m * stride + hi;
  v16h out;
#pragma unroll
  for (int i = 0; i < 8; ++i) {
    out[i]     = (_Float16)r[i];
    out[i + 8] = (_Float16)r[i + 16];
  }
  return out;
}

// B (32x16, KxN), source stored as [n][k] row-major with given stride:
// lane column n = lane&15; halves e -> K = (e>>3)*16 + (lane>>4)*8 + (e&7)
__device__ __forceinline__ v16h load_frag_b(const _Float16* base, int stride) {
  const int lane = threadIdx.x & 31;
  const int n  = lane & 15;
  const int k0 = (lane >> 4) * 8;
  const _Float16* r = base + n * stride + k0;
  v8h lo = *(const v8h*)(r);
  v8h hh = *(const v8h*)(r + 16);
  v16h out;
#pragma unroll
  for (int i = 0; i < 8; ++i) { out[i] = lo[i]; out[i + 8] = hh[i]; }
  return out;
}

// ---- sizes ----------------------------------------------------------------
#define NB   16
#define LQ   256
#define EMB  1024
#define SS   2
#define HH   16
#define DD   64

// ---------------------------------------------------------------------------
// Kernel 0: Wo f32 -> f16
__global__ void k_cvt_wo(const float* __restrict__ Wo, _Float16* __restrict__ Wo16) {
  int i = blockIdx.x * 256 + threadIdx.x;
  if (i < EMB * EMB) Wo16[i] = (_Float16)Wo[i];
}

// ---------------------------------------------------------------------------
// Kernel 1: per-head projections.  One block = (n, s, h, which).
// Y(256x64) = X(256x64) @ W(64x64)^T  via WMMA.  V stored transposed [d][l].
__global__ __launch_bounds__(128) void k_project(
    const float* __restrict__ vals, const float* __restrict__ keys_,
    const float* __restrict__ qry,
    const float* __restrict__ Wv, const float* __restrict__ Wk,
    const float* __restrict__ Wq,
    _Float16* __restrict__ Qp, _Float16* __restrict__ Kp,
    _Float16* __restrict__ Vt) {
  __shared__ _Float16 __attribute__((aligned(32))) sW[DD * DD];     // [e][d]
  __shared__ _Float16 __attribute__((aligned(32))) sX[LQ * DD];     // [l][d]

  int b = blockIdx.x;
  const int which = b % 3;  b /= 3;          // 0=q 1=k 2=v
  const int h = b % HH;     b /= HH;
  const int s = b % SS;
  const int n = b / SS;

  const float* src = (which == 0) ? qry : (which == 1) ? keys_ : vals;
  const float* W   = (which == 0) ? Wq  : (which == 1) ? Wk    : Wv;

  const int tid = threadIdx.x;
  // stage W (64x64) as f16
  for (int i = tid; i < DD * DD; i += 128) sW[i] = (_Float16)W[i];
  // stage X (256x64) as f16; src layout (n, l, E, S)
  for (int i = tid; i < LQ * DD; i += 128) {
    int l = i >> 6, d = i & 63;
    float x = src[(((size_t)n * LQ + l) * EMB + h * DD + d) * SS + s];
    sX[i] = (_Float16)x;
  }
  __syncthreads();

  const int wave = tid >> 5;            // 4 waves -> 4 ntiles (D = 64)
  const int lane = tid & 31;
  const int nt = wave;
  // B fragments fixed per wave
  v16h b0 = load_frag_b(sW + nt * 16 * DD + 0,  DD);
  v16h b1 = load_frag_b(sW + nt * 16 * DD + 32, DD);

  const size_t slab = ((((size_t)n * HH + h) * SS + s));
  _Float16* qk_out = (which == 0) ? (Qp + slab * LQ * DD)
                                  : (Kp + slab * LQ * DD);
  _Float16* v_out  = Vt + slab * DD * LQ;

  for (int mt = 0; mt < 16; ++mt) {
    v16h a0 = load_frag_a(sX + mt * 16 * DD + 0,  DD);
    v16h a1 = load_frag_a(sX + mt * 16 * DD + 32, DD);
    v8f c = {};
    c = WMMA_F16(a0, b0, c);
    c = WMMA_F16(a1, b1, c);
    const int col = lane & 15;
    const int rbase = mt * 16 + ((lane >> 4) * 8);
#pragma unroll
    for (int i = 0; i < 8; ++i) {
      int row = rbase + i;                 // L index
      int d   = nt * 16 + col;             // D index
      if (which == 2) v_out[d * LQ + row]  = (_Float16)c[i];   // transposed
      else            qk_out[row * DD + d] = (_Float16)c[i];
    }
  }
}

// ---------------------------------------------------------------------------
// Kernel 2: attention for one (n,h,s) slice.  Full 256x256 energy in LDS
// (256 KB — uses the CDNA5 320 KB WGP LDS), column softmax (axis = q),
// then Ctx = Att @ V, stored directly in torch-reshaped (n,L,S,1024) layout.
__global__ __launch_bounds__(256) void k_attn(
    const _Float16* __restrict__ Qp, const _Float16* __restrict__ Kp,
    const _Float16* __restrict__ Vt, const int* __restrict__ maskp,
    _Float16* __restrict__ CtxB) {
  extern __shared__ char smem[];
  float* EN  = (float*)smem;                       // [256][256]
  int*   MSK = (int*)(smem + LQ * LQ * sizeof(float));

  const int bid = blockIdx.x;
  const int s = bid & 1;
  const int h = (bid >> 1) & 15;
  const int n = bid >> 5;
  const int tid = threadIdx.x;
  const int wave = tid >> 5;
  const int lane = tid & 31;

  // faithful replication of the scrambled torch mask reshape
  {
    const int q = tid;                                      // 256 threads
    const int idx = h * 512 + s * 256 + q;                  // (H,S,L) flat
    const int so  = idx >> 12;
    const int lo  = (idx & 4095) >> 4;
    MSK[q] = maskp[((size_t)n * SS + so) * (EMB * LQ) + lo];
  }
  __syncthreads();

  const size_t slab = (((size_t)n * HH + h) * SS + s);
  const _Float16* Qb = Qp + slab * LQ * DD;
  const _Float16* Kb = Kp + slab * LQ * DD;
  const _Float16* Vb = Vt + slab * DD * LQ;

  // ---- energy = Q @ K^T, scaled + masked into LDS ----
  const float inv = 0.03125f;  // 1/sqrt(1024)
  for (int mi = 0; mi < 2; ++mi) {
    const int mt = wave * 2 + mi;
    v16h a0 = load_frag_a(Qb + mt * 16 * DD + 0,  DD);
    v16h a1 = load_frag_a(Qb + mt * 16 * DD + 32, DD);
    for (int nt = 0; nt < 16; ++nt) {
      v16h b0 = load_frag_b(Kb + nt * 16 * DD + 0,  DD);
      v16h b1 = load_frag_b(Kb + nt * 16 * DD + 32, DD);
      v8f c = {};
      c = WMMA_F16(a0, b0, c);
      c = WMMA_F16(a1, b1, c);
      const int col = nt * 16 + (lane & 15);
      const int rbase = mt * 16 + ((lane >> 4) * 8);
#pragma unroll
      for (int i = 0; i < 8; ++i) {
        int q = rbase + i;
        EN[q * LQ + col] = MSK[q] ? c[i] * inv : -1e20f;
      }
    }
  }
  __syncthreads();

  // ---- softmax over q (axis 3) : one COLUMN per thread ----
  {
    const int l = tid;
    float mx = -3.4e38f;
    for (int q = 0; q < LQ; ++q) { float v = EN[q * LQ + l]; mx = v > mx ? v : mx; }
    float sm = 0.f;
    for (int q = 0; q < LQ; ++q) sm += __expf(EN[q * LQ + l] - mx);
    const float r = 1.f / sm;
    for (int q = 0; q < LQ; ++q)
      EN[q * LQ + l] = __expf(EN[q * LQ + l] - mx) * r;
  }
  __syncthreads();

  // ---- Ctx = Att @ V ----
  for (int mi = 0; mi < 2; ++mi) {
    const int mt = wave * 2 + mi;
    v8f acc[4] = {};
    for (int kt = 0; kt < 8; ++kt) {
      v16h a = load_frag_a_f32(EN + mt * 16 * LQ + kt * 32, LQ);
#pragma unroll
      for (int nt = 0; nt < 4; ++nt) {
        v16h b = load_frag_b(Vb + nt * 16 * LQ + kt * 32, LQ);
        acc[nt] = WMMA_F16(a, b, acc[nt]);
      }
    }
    const int col = lane & 15;
    const int rbase = mt * 16 + ((lane >> 4) * 8);
#pragma unroll
    for (int nt = 0; nt < 4; ++nt) {
      const int d = nt * 16 + col;
      const int idx = h * 128 + d * 2 + s;     // torch reshape (q,H,D,S)->(q,S,E)
      const int sp = idx >> 10, ep = idx & 1023;
#pragma unroll
      for (int i = 0; i < 8; ++i) {
        int q = rbase + i;
        CtxB[((((size_t)n * LQ + q) * SS) + sp) * EMB + ep] = (_Float16)acc[nt][i];
      }
    }
  }
}

// ---------------------------------------------------------------------------
// Kernel 3: output projection  Y(16384x1024) = Ctx @ Wo^T + bo  (dominant GEMM)
// The 64x1024 Wo panel (128 KB) is staged into LDS once per block via the
// Tensor Data Mover (TDM), then all 8 waves read B fragments from LDS.
__global__ __launch_bounds__(256) void k_outproj(
    const _Float16* __restrict__ CtxB, const _Float16* __restrict__ Wo16,
    const float* __restrict__ bo, float* __restrict__ out) {
  extern __shared__ char smemB[];                 // 64 * 1024 f16 = 128 KB
  _Float16* sB = (_Float16*)smemB;                // panel [f_rel][e]

  const int bm = blockIdx.x >> 4;     // 128 M-blocks (x 8 waves x 16 rows)
  const int bn = blockIdx.x & 15;     // 16 N-blocks of 64
  const int tid  = threadIdx.x;
  const int wave = tid >> 5;
  const int lane = tid & 31;

  const _Float16* wsrc = Wo16 + (size_t)(bn * 64) * EMB;   // panel origin

#if defined(HAVE_TDM)
  if (wave == 0) {   // TDM ignores EXEC; issue from one wave (uniform branch)
    // D# per CDNA5 ISA: 2D tile 1024x64, 2-byte elems,
    // contiguous LDS dest at offset 0 (sole/dynamic LDS object).
    const unsigned long long ga = (unsigned long long)wsrc;
    v4u g0;
    g0[0] = 1u;                                       // count=1, user mode
    g0[1] = 0u;                                       // lds_addr = 0
    g0[2] = (unsigned)(ga & 0xFFFFFFFFu);             // global_addr[31:0]
    g0[3] = (unsigned)(((ga >> 32) & 0x01FFFFFFu) | (2u << 30)); // [56:32]|type=2
    v8i g1;
    g1[0] = 0x00010000;        // workgroup_mask=0, data_size=1 (2 bytes)
    g1[1] = (int)(1024u << 16);        // tensor_dim0[15:0] = 1024
    g1[2] = (int)(64u << 16);          // tensor_dim0 hi=0 | tensor_dim1 = 64
    g1[3] = (int)(1024u << 16);        // tensor_dim1 hi=0 | tile_dim0 = 1024
    g1[4] = 64;                        // tile_dim1 = 64, tile_dim2 = 0
    g1[5] = 1024;                      // tensor_dim0_stride[31:0] = 1024
    g1[6] = 0;                         // stride hi | tensor_dim1_stride lo
    g1[7] = 0;
    v4i g2 = {0, 0, 0, 0};
    v4i g3 = {0, 0, 0, 0};
#if __has_include(<hip/amd_detail/amd_gfx1250_TDM.h>)
    v8i gz = {0, 0, 0, 0, 0, 0, 0, 0};          // clang-23 / therock: 6-arg form
    __builtin_amdgcn_tensor_load_to_lds(g0, g1, g2, g3, gz, 0);
#else
    __builtin_amdgcn_tensor_load_to_lds(g0, g1, g2, g3, 0);   // ROCm 7.2: 5-arg
#endif
    __builtin_amdgcn_s_wait_tensorcnt((unsigned short)0);
  }
  __syncthreads();
#else
  // fallback: cooperative vector copy (also used by the host compile pass)
  for (int i = tid * 8; i < 64 * EMB; i += 256 * 8)
    *(v8h*)(sB + i) = *(const v8h*)(wsrc + i);
  __syncthreads();
#endif

  const int r0 = (bm * 8 + wave) * 16;
  const _Float16* arow = CtxB + (size_t)r0 * EMB;

  v8f acc[4] = {};
  for (int kt = 0; kt < 32; ++kt) {
    const int kb = kt * 32;
    if (kt + 1 < 32) __builtin_prefetch(arow + kb + 32, 0, 0);  // global_prefetch_b8
    v16h a = load_frag_a(arow + kb, EMB);
#pragma unroll
    for (int nt = 0; nt < 4; ++nt) {
      v16h b = load_frag_b(sB + (size_t)(nt * 16) * EMB + kb, EMB);  // from LDS
      acc[nt] = WMMA_F16(a, b, acc[nt]);
    }
  }

  const int col = lane & 15;
  const int rb  = r0 + ((lane >> 4) * 8);
#pragma unroll
  for (int nt = 0; nt < 4; ++nt) {
    const int f = bn * 64 + nt * 16 + col;
    const float bf = bo[f];
#pragma unroll
    for (int i = 0; i < 8; ++i)
      out[(size_t)(rb + i) * EMB + f] = acc[nt][i] + bf;
  }
}

// ---------------------------------------------------------------------------
extern "C" void kernel_launch(void* const* d_in, const int* in_sizes, int n_in,
                              void* d_out, int out_size, void* d_ws, size_t ws_size,
                              hipStream_t stream) {
  const float* vals  = (const float*)d_in[0];
  const float* keys_ = (const float*)d_in[1];
  const float* qry   = (const float*)d_in[2];
  const int*   msk   = (const int*)d_in[3];
  const float* Wv    = (const float*)d_in[4];
  const float* Wk    = (const float*)d_in[5];
  const float* Wq    = (const float*)d_in[6];
  const float* Wo    = (const float*)d_in[7];
  const float* bo    = (const float*)d_in[8];
  float* out = (float*)d_out;

  // workspace layout (all f16): Qp/Kp/Vt 16 MB each, CtxB 16 MB, Wo16 2 MB
  char* w = (char*)d_ws;
  const size_t SLAB = (size_t)NB * HH * SS * LQ * DD;   // 8.4M halves
  _Float16* Qp   = (_Float16*)(w);
  _Float16* Kp   = (_Float16*)(w + SLAB * 2);
  _Float16* Vt   = (_Float16*)(w + SLAB * 4);
  _Float16* CtxB = (_Float16*)(w + SLAB * 6);
  _Float16* Wo16 = (_Float16*)(w + SLAB * 8);

  k_cvt_wo<<<(EMB * EMB) / 256, 256, 0, stream>>>(Wo, Wo16);

  k_project<<<NB * SS * HH * 3, 128, 0, stream>>>(vals, keys_, qry,
                                                  Wv, Wk, Wq, Qp, Kp, Vt);

  const size_t attn_lds = (size_t)LQ * LQ * sizeof(float) + LQ * sizeof(int);
  k_attn<<<NB * HH * SS, 256, attn_lds, stream>>>(Qp, Kp, Vt, msk, CtxB);

  const size_t panel_lds = (size_t)64 * EMB * sizeof(_Float16);   // 128 KB
  k_outproj<<<128 * 16, 256, panel_lds, stream>>>(CtxB, Wo16, bo, out);
}
